// LSTM_BN_64991445123575
// MI455X (gfx1250) — matrix operand order repaired and checked by reading, not compiled, not run on hardware
//
#include <hip/hip_runtime.h>
#include <hip/hip_bf16.h>
#include <math.h>

typedef __attribute__((ext_vector_type(16))) _Float16 v16h;
typedef __attribute__((ext_vector_type(8)))  _Float16 v8h;
typedef __attribute__((ext_vector_type(8)))  float    v8f;

#define B_    16
#define T_    2048
#define IN_   65
#define WIN_  12
#define CTX_  25
#define K0_   1625          // CTX_*IN_
#define K0P_  1632          // padded to multiple of 32 (51 k-steps)
#define HID_  64
#define G_    256           // 4*HID_
#define NROW_ (B_*T_)       // 32768
#define PADT_ (T_ + 2*WIN_) // 2072
#define SLAB_ (PADT_*IN_)   // 134680 halves per batch row

union F16x16 { v16h v; struct { v8h lo; v8h hi; } p; };

// A fragment (16x32 f16, row-major source, ld in halves)
__device__ __forceinline__ v16h load_frag_A(const _Float16* src, int ld, int k0) {
  int lane = threadIdx.x & 31;
  int row  = lane & 15;
  int kb   = k0 + ((lane & 16) ? 8 : 0);
  const _Float16* p = src + row * ld + kb;
  F16x16 r;
  r.p.lo = *(const v8h*)(p);
  r.p.hi = *(const v8h*)(p + 16);
  return r.v;
}

// B fragment (32x16 f16), weight stored N x K row-major
__device__ __forceinline__ v16h load_frag_B(const _Float16* wNK, int ld, int n0, int k0) {
  int lane = threadIdx.x & 31;
  int col  = lane & 15;
  int kb   = k0 + ((lane & 16) ? 8 : 0);
  const _Float16* p = wNK + (size_t)(n0 + col) * ld + kb;
  F16x16 r;
  r.p.lo = *(const v8h*)(p);
  r.p.hi = *(const v8h*)(p + 16);
  return r.v;
}

__device__ __forceinline__ v8f wmma_(v16h a, v16h b, v8f c) {
  return __builtin_amdgcn_wmma_f32_16x16x32_f16(false, a, false, b, (short)0, c, false, false);
}

__device__ __forceinline__ float fast_tanhf(float x) {
#if __has_builtin(__builtin_amdgcn_tanhf)
  return __builtin_amdgcn_tanhf(x);
#else
  return tanhf(x);
#endif
}
__device__ __forceinline__ float sigmoid_(float x) {
  return fmaf(fast_tanhf(x * 0.5f), 0.5f, 0.5f);
}

// ---------------- input packing: f32 -> f16 with +/-12 frame zero pad ----------------
// inp16[b][tp][d] = input[b][tp-12][d] (zero outside). A row of the concat matrix is
// then CONTIGUOUS: xcat[b,t,k] = inp16[b*SLAB_ + t*65 + k].
__global__ void pack_in16_kernel(const float* __restrict__ in, _Float16* __restrict__ inp16) {
  int idx = blockIdx.x * blockDim.x + threadIdx.x;
  int total = B_ * SLAB_ + 64; // 64-half zero tail for padded-K overread
  if (idx >= total) return;
  if (idx >= B_ * SLAB_) { inp16[idx] = (_Float16)0.0f; return; }
  int b  = idx / SLAB_;
  int rr = idx - b * SLAB_;
  int tp = rr / IN_;
  int d  = rr - tp * IN_;
  float v = 0.0f;
  int t = tp - WIN_;
  if (t >= 0 && t < T_) v = in[((size_t)(b * T_ + t)) * IN_ + d];
  inp16[idx] = (_Float16)v;
}

// ---------------- weight packing ----------------
__global__ void pack_w0_kernel(const float* __restrict__ Wih0, _Float16* __restrict__ W0p) {
  int idx = blockIdx.x * blockDim.x + threadIdx.x;
  int total = G_ * K0P_;
  if (idx >= total) return;
  int g = idx / K0P_;
  int k = idx - g * K0P_;
  float v = (k < K0_) ? Wih0[g * K0_ + k] : 0.0f;
  W0p[idx] = (_Float16)v;
}

__global__ void pack_small_kernel(const float* __restrict__ Whh0, const float* __restrict__ Wih1,
                                  const float* __restrict__ Whh1, const float* __restrict__ Wih2,
                                  const float* __restrict__ Whh2,
                                  const float* __restrict__ bih0, const float* __restrict__ bhh0,
                                  const float* __restrict__ bih1, const float* __restrict__ bhh1,
                                  const float* __restrict__ bih2, const float* __restrict__ bhh2,
                                  _Float16* __restrict__ Whh0p, _Float16* __restrict__ Wih1p,
                                  _Float16* __restrict__ Whh1p, _Float16* __restrict__ Wih2p,
                                  _Float16* __restrict__ Whh2p, float* __restrict__ bsum,
                                  int* __restrict__ prog) {
  int idx = blockIdx.x * blockDim.x + threadIdx.x;
  const int WSZ = G_ * HID_; // 16384
  if (idx < WSZ)            Whh0p[idx]          = (_Float16)Whh0[idx];
  else if (idx < 2 * WSZ)   Wih1p[idx - WSZ]    = (_Float16)Wih1[idx - WSZ];
  else if (idx < 3 * WSZ)   Whh1p[idx - 2*WSZ]  = (_Float16)Whh1[idx - 2*WSZ];
  else if (idx < 4 * WSZ)   Wih2p[idx - 3*WSZ]  = (_Float16)Wih2[idx - 3*WSZ];
  else if (idx < 5 * WSZ)   Whh2p[idx - 4*WSZ]  = (_Float16)Whh2[idx - 4*WSZ];
  else {
    int r = idx - 5 * WSZ;
    if (r < 3 * G_) {
      int l = r >> 8, n = r & 255;
      const float* bi = (l == 0) ? bih0 : ((l == 1) ? bih1 : bih2);
      const float* bh = (l == 0) ? bhh0 : ((l == 1) ? bhh1 : bhh2);
      bsum[r] = bi[n] + bh[n];
    } else if (r < 3 * G_ + 8) {
      prog[r - 3 * G_] = 0; // reset pipeline progress flags every launch
    }
  }
}

// ---------------- layer-0 windowed input GEMM (compute-bound, 27 GFLOP) ----------------
// pre[r, g] = xcat[r, :] . Wih0[g, :] + bih0[g] + bhh0[g]
// A tiles copied contiguously from padded f16 input; double-buffered LDS staging.
__global__ void __launch_bounds__(256) gemm_pre0_kernel(const _Float16* __restrict__ inp16,
                                                        const _Float16* __restrict__ W0p,
                                                        const float* __restrict__ bsum,
                                                        float* __restrict__ pre) {
  __shared__ __align__(16) _Float16 At[2][16][40];
  int tid  = threadIdx.x;
  int lane = tid & 31;
  int wave = tid >> 5;           // 8 waves x 2 N-tiles
  int rowBase = blockIdx.x * 16; // 2048 M-tiles
  int b  = rowBase >> 11;
  int t0 = rowBase & 2047;
  int n0 = wave * 32;
  const _Float16* src = inp16 + (size_t)b * SLAB_ + (size_t)t0 * IN_;
  int r0 = tid >> 5, c0 = tid & 31; // each thread stages 2 halves, branch-free

  At[0][r0][c0]     = src[r0 * IN_ + c0];
  At[0][r0 + 8][c0] = src[(r0 + 8) * IN_ + c0];
  __syncthreads();

  v8f acc0 = {}; v8f acc1 = {};
  for (int ks = 0; ks < K0P_ / 32; ++ks) {
    int cur = ks & 1;
    if (ks + 1 < K0P_ / 32) { // prefetch next k-slice into the other buffer
      int kofs = (ks + 1) * 32;
      At[cur ^ 1][r0][c0]     = src[r0 * IN_ + kofs + c0];
      At[cur ^ 1][r0 + 8][c0] = src[(r0 + 8) * IN_ + kofs + c0];
    }
    v16h a  = load_frag_A(&At[cur][0][0], 40, 0);
    v16h b0 = load_frag_B(W0p, K0P_, n0, ks * 32);
    v16h b1 = load_frag_B(W0p, K0P_, n0 + 16, ks * 32);
    acc0 = wmma_(a, b0, acc0);
    acc1 = wmma_(a, b1, acc1);
    __syncthreads();
  }
  int mb = (lane & 16) ? 8 : 0;
  int nc = lane & 15;
#pragma unroll
  for (int v = 0; v < 8; ++v) {
    int m = mb + v;
    pre[(size_t)(rowBase + m) * G_ + n0 + nc]      = acc0[v] + bsum[n0 + nc];
    pre[(size_t)(rowBase + m) * G_ + n0 + 16 + nc] = acc1[v] + bsum[n0 + 16 + nc];
  }
}

// ---------------- pipelined 3-layer recurrent scan ----------------
// Block l = layer l; 16 waves x 1 N-tile each (shortest WMMA dependency chain).
// Layer 0 consumes precomputed pre0; layers 1/2 fuse their input GEMM
// (h_in[t] @ Wih^T) into the step and chase layer l-1 with a 1-step lag via an
// acquire/release progress flag. Weight B-fragments stay resident in VGPRs, and the
// recurrent/input products accumulate in independent chains merged by a vector add.
__global__ void __launch_bounds__(512) lstm_pipeline_kernel(
    const float* __restrict__ pre0,
    const _Float16* __restrict__ Whh0p, const _Float16* __restrict__ Wih1p,
    const _Float16* __restrict__ Whh1p, const _Float16* __restrict__ Wih2p,
    const _Float16* __restrict__ Whh2p,
    const float* __restrict__ bsum,
    _Float16* __restrict__ h0, _Float16* __restrict__ h1, _Float16* __restrict__ h2,
    int* __restrict__ prog) {
  __shared__ __align__(16) float    gls[16][G_];
  __shared__ __align__(16) float    cls[16][HID_];
  __shared__ __align__(16) _Float16 h16[16][HID_];
  int l   = blockIdx.x;
  int tid = threadIdx.x, lane = tid & 31, wave = tid >> 5;
  int n0  = wave * 16; // 16 waves, one 16x16 N-tile each
  const _Float16* WhhP = (l == 0) ? Whh0p : ((l == 1) ? Whh1p : Whh2p);
  const _Float16* WihP = (l == 1) ? Wih1p : Wih2p;
  const _Float16* hin  = (l == 1) ? h0 : h1;
  _Float16* hout = (l == 0) ? h0 : ((l == 1) ? h1 : h2);
  const float* bs = bsum + l * G_;

  for (int p = tid; p < 16 * HID_; p += 512) {
    cls[p >> 6][p & 63] = 0.0f;
    h16[p >> 6][p & 63] = (_Float16)0.0f;
  }
  v16h Bh0 = load_frag_B(WhhP, HID_, n0, 0);
  v16h Bh1 = load_frag_B(WhhP, HID_, n0, 32);
  v16h Bi0 = {}, Bi1 = {};
  if (l > 0) {
    Bi0 = load_frag_B(WihP, HID_, n0, 0);
    Bi1 = load_frag_B(WihP, HID_, n0, 32);
  }
  __syncthreads();

  int mb = (lane & 16) ? 8 : 0;
  int nc = lane & 15;
  for (int t = 0; t < T_; ++t) {
    if (l > 0) {
      if (tid == 0) {
        while (__hip_atomic_load(&prog[l - 1], __ATOMIC_ACQUIRE,
                                 __HIP_MEMORY_SCOPE_AGENT) <= t)
          __builtin_amdgcn_s_sleep(1);
      }
      __syncthreads();
    }
    v16h a0 = load_frag_A(&h16[0][0], HID_, 0);
    v16h a1 = load_frag_A(&h16[0][0], HID_, 32);
    v8f accH = {};
    accH = wmma_(a0, Bh0, accH);
    accH = wmma_(a1, Bh1, accH);
    if (l > 0) {
      const _Float16* hb = hin + (size_t)t * (16 * HID_); // t-major [t][m][j]
      v16h ai0 = load_frag_A(hb, HID_, 0);
      v16h ai1 = load_frag_A(hb, HID_, 32);
      v8f accI = {};
      accI = wmma_(ai0, Bi0, accI); // independent chain: runs parallel to accH
      accI = wmma_(ai1, Bi1, accI);
#pragma unroll
      for (int v = 0; v < 8; ++v)
        gls[mb + v][n0 + nc] = accH[v] + accI[v] + bs[n0 + nc];
    } else {
#pragma unroll
      for (int v = 0; v < 8; ++v) {
        int m = mb + v; // m == batch index
        gls[m][n0 + nc] = accH[v] + pre0[((size_t)m * T_ + t) * G_ + n0 + nc];
      }
      if (t + 1 < T_) { // hint next step's pre rows toward the caches
        __builtin_prefetch(&pre0[((size_t)mb * T_ + t + 1) * G_ + n0 + nc], 0, 1);
        __builtin_prefetch(&pre0[((size_t)(mb + 4) * T_ + t + 1) * G_ + n0 + nc], 0, 1);
      }
    }
    __syncthreads();
#pragma unroll
    for (int q = 0; q < 2; ++q) {
      int p = tid + 512 * q; // 1024 (m,j) pairs over 512 threads
      int m = p >> 6, j = p & 63;
      float gi = gls[m][j];
      float gf = gls[m][j + 64];
      float gg = gls[m][j + 128];
      float go = gls[m][j + 192];
      float cc = sigmoid_(gf) * cls[m][j] + sigmoid_(gi) * fast_tanhf(gg);
      float hh = sigmoid_(go) * fast_tanhf(cc);
      cls[m][j] = cc;
      _Float16 hf = (_Float16)hh;
      h16[m][j] = hf;
      hout[(size_t)t * (16 * HID_) + m * HID_ + j] = hf;
    }
    if (l < 2) { // layer 2 has no in-kernel consumer; kernel boundary orders h2
      __threadfence();
      __syncthreads();
      if (tid == 0)
        __hip_atomic_store(&prog[l], t + 1, __ATOMIC_RELEASE, __HIP_MEMORY_SCOPE_AGENT);
    } else {
      __syncthreads();
    }
  }
}

// ---------------- output head: sigmoid(relu(h) @ Wout + bout) ----------------
__global__ void output_kernel(const _Float16* __restrict__ h2, const float* __restrict__ Wout,
                              const float* __restrict__ bout, float* __restrict__ out) {
  int idx = blockIdx.x * blockDim.x + threadIdx.x;
  if (idx >= NROW_) return;
  int t = idx >> 4;
  int m = idx & 15;
  float acc = bout[0];
#pragma unroll
  for (int j = 0; j < HID_; ++j) {
    float v = (float)h2[(size_t)idx * HID_ + j]; // idx*64 == t*1024 + m*64
    acc += fmaxf(v, 0.0f) * Wout[j];
  }
  out[(size_t)m * T_ + t] = sigmoid_(acc);
}

extern "C" void kernel_launch(void* const* d_in, const int* in_sizes, int n_in,
                              void* d_out, int out_size, void* d_ws, size_t ws_size,
                              hipStream_t stream) {
  const float* in_audio = (const float*)d_in[0];
  const float* Wih0 = (const float*)d_in[1];
  const float* Whh0 = (const float*)d_in[2];
  const float* bih0 = (const float*)d_in[3];
  const float* bhh0 = (const float*)d_in[4];
  const float* Wih1 = (const float*)d_in[5];
  const float* Whh1 = (const float*)d_in[6];
  const float* bih1 = (const float*)d_in[7];
  const float* bhh1 = (const float*)d_in[8];
  const float* Wih2 = (const float*)d_in[9];
  const float* Whh2 = (const float*)d_in[10];
  const float* bih2 = (const float*)d_in[11];
  const float* bhh2 = (const float*)d_in[12];
  const float* Wout = (const float*)d_in[13];
  const float* bout = (const float*)d_in[14];
  float* out = (float*)d_out;

  char* ws = (char*)d_ws;
  const size_t OFF_W0P  = 0;
  const size_t OFF_WHH0 = OFF_W0P  + (size_t)G_ * K0P_ * 2;   // 835584
  const size_t OFF_WIH1 = OFF_WHH0 + (size_t)G_ * HID_ * 2;
  const size_t OFF_WHH1 = OFF_WIH1 + (size_t)G_ * HID_ * 2;
  const size_t OFF_WIH2 = OFF_WHH1 + (size_t)G_ * HID_ * 2;
  const size_t OFF_WHH2 = OFF_WIH2 + (size_t)G_ * HID_ * 2;
  const size_t OFF_BSUM = OFF_WHH2 + (size_t)G_ * HID_ * 2;
  const size_t OFF_PROG = OFF_BSUM + (size_t)3 * G_ * 4;
  const size_t OFF_IN16 = OFF_PROG + 256;
  const size_t OFF_PRE  = OFF_IN16 + 4310016;                  // (B_*SLAB_+64)*2 padded
  const size_t OFF_H0   = OFF_PRE  + (size_t)NROW_ * G_ * 4;
  const size_t OFF_H1   = OFF_H0   + (size_t)NROW_ * HID_ * 2;
  const size_t OFF_H2   = OFF_H1   + (size_t)NROW_ * HID_ * 2;
  // total ~= 49 MB

  _Float16* W0p   = (_Float16*)(ws + OFF_W0P);
  _Float16* Whh0p = (_Float16*)(ws + OFF_WHH0);
  _Float16* Wih1p = (_Float16*)(ws + OFF_WIH1);
  _Float16* Whh1p = (_Float16*)(ws + OFF_WHH1);
  _Float16* Wih2p = (_Float16*)(ws + OFF_WIH2);
  _Float16* Whh2p = (_Float16*)(ws + OFF_WHH2);
  float*    bsum  = (float*)(ws + OFF_BSUM);
  int*      prog  = (int*)(ws + OFF_PROG);
  _Float16* inp16 = (_Float16*)(ws + OFF_IN16);
  float*    pre   = (float*)(ws + OFF_PRE);
  _Float16* h0    = (_Float16*)(ws + OFF_H0);
  _Float16* h1    = (_Float16*)(ws + OFF_H1);
  _Float16* h2    = (_Float16*)(ws + OFF_H2);

  pack_w0_kernel<<<(G_ * K0P_ + 255) / 256, 256, 0, stream>>>(Wih0, W0p);
  pack_small_kernel<<<(5 * G_ * HID_ + 3 * G_ + 8 + 255) / 256, 256, 0, stream>>>(
      Whh0, Wih1, Whh1, Wih2, Whh2, bih0, bhh0, bih1, bhh1, bih2, bhh2,
      Whh0p, Wih1p, Whh1p, Wih2p, Whh2p, bsum, prog);
  pack_in16_kernel<<<(B_ * SLAB_ + 64 + 255) / 256, 256, 0, stream>>>(in_audio, inp16);

  gemm_pre0_kernel<<<NROW_ / 16, 256, 0, stream>>>(inp16, W0p, bsum + 0 * G_, pre);

  lstm_pipeline_kernel<<<3, 512, 0, stream>>>(pre, Whh0p, Wih1p, Whh1p, Wih2p, Whh2p,
                                              bsum, h0, h1, h2, prog);

  output_kernel<<<(NROW_ + 255) / 256, 256, 0, stream>>>(h2, Wout, bout, out);
}
